// TriX6502GeometriX_65884798321351
// MI455X (gfx1250) — compile-verified, matrix operand order checked by reading, not run on hardware
//
#include <hip/hip_runtime.h>
#include <hip/hip_bf16.h>
#include <math.h>

#define D_ 512
#define E_ 16
#define K_ 4
#define H_ 1024
#define MT 32          // tokens per MoE block (two 16-row WMMA M-tiles sharing staged weights)
#define SPREAD2X 4.5f  // 2 * 1.5^2

typedef __attribute__((ext_vector_type(16))) __bf16 v16bf;
typedef __attribute__((ext_vector_type(8)))  __bf16 v8bf;
typedef __attribute__((ext_vector_type(4)))  __bf16 v4bf;
typedef __attribute__((ext_vector_type(8)))  float  v8f;

union ABFrag { v16bf v; v8bf h[2]; };

// ---- XOR-swizzled LDS addressing (16B granules): conflict-free fragment reads, zero padding
__device__ __forceinline__ char* swzp(char* base, int row, int rowBytes, int byteInRow, int cmask) {
  int ch = (byteInRow >> 4) ^ (row & cmask);
  return base + row * rowBytes + (ch << 4) + (byteInRow & 15);
}

// ---- async global->LDS stage of a [512 rows x 32 K] bf16 tile (64B/row, swizzled), 256 threads
__device__ __forceinline__ void stage_bt(char* btb, const __bf16* wbase, int rowStrideElems) {
  int t = threadIdx.x;
  int chunk = t & 3;         // 16B chunk within row
  int rsub  = t >> 2;        // 0..63
#pragma unroll
  for (int j = 0; j < 8; ++j) {
    int r = j * 64 + rsub;
    const __bf16* g = wbase + (size_t)r * rowStrideElems + chunk * 8;
    unsigned lds = (unsigned)(size_t)swzp(btb, r, 64, chunk * 16, 3);
    asm volatile("global_load_async_to_lds_b128 %0, %1, off"
                 :: "v"(lds), "v"((unsigned long long)(size_t)g)
                 : "memory");
  }
}

// ================= kernel 1: weight convert/transpose + zero counters =================
__global__ void trix_prep(const float* __restrict__ W1, const float* __restrict__ W2,
                          __bf16* __restrict__ W1t, __bf16* __restrict__ W2t,
                          int* __restrict__ cnt, float* __restrict__ usage_out) {
  const long long N1 = (long long)E_ * D_ * H_;
  long long stride = (long long)gridDim.x * blockDim.x;
  for (long long i = (long long)blockIdx.x * blockDim.x + threadIdx.x; i < 2 * N1; i += stride) {
    if (i < N1) {               // W1t[e][h][d] = W1[e][d][h]  (rows = output col h, inner = K d)
      long long e = i / ((long long)H_ * D_);
      long long r = i % ((long long)H_ * D_);
      long long h = r / D_, d = r % D_;
      W1t[i] = (__bf16)W1[(e * D_ + d) * H_ + h];
    } else {                    // W2t[e][d][h] = W2[e][h][d]  (rows = output col d, inner = K h)
      long long j = i - N1;
      long long e = j / ((long long)D_ * H_);
      long long r = j % ((long long)D_ * H_);
      long long d = r / H_, h = r % H_;
      W2t[j] = (__bf16)W2[(e * H_ + h) * D_ + d];
    }
  }
  if (blockIdx.x == 0 && threadIdx.x < E_) {
    cnt[threadIdx.x] = 0;
    usage_out[threadIdx.x] = 0.f;
  }
}

// ================= kernel 2: features -> x, router, top-4 gates, expert bucketing =================
__global__ void __launch_bounds__(128)
trix_front(const int* __restrict__ op_idx, const int* __restrict__ a,
           const int* __restrict__ b, const int* __restrict__ c,
           const float* __restrict__ op_embed_w, const float* __restrict__ proj_w,
           const float* __restrict__ proj_b, const float* __restrict__ router_w,
           const float* __restrict__ router_b, const float* __restrict__ tile_centers,
           float* __restrict__ xbuf, float* __restrict__ oacc,
           float* __restrict__ gates_out, float* __restrict__ usage_out,
           int* __restrict__ cnt, int* __restrict__ list, float* __restrict__ glist, int B) {
  __shared__ float feat[49];
  __shared__ float xs[D_];
  __shared__ float lg[E_];
  int bt = blockIdx.x, t = threadIdx.x;
  int op = op_idx[bt];
  if (t < 32) feat[t] = op_embed_w[op * 32 + t];
  else if (t < 40) feat[t] = (float)((a[bt] >> (t - 32)) & 1);
  else if (t < 48) feat[t] = (float)((b[bt] >> (t - 40)) & 1);
  else if (t == 48) feat[48] = (float)c[bt];
  __syncthreads();
#pragma unroll
  for (int j = 0; j < 4; ++j) {
    int d = t + 128 * j;
    float acc = proj_b[d];
    for (int f = 0; f < 49; ++f) acc += feat[f] * proj_w[f * D_ + d];
    xs[d] = acc;
    xbuf[(size_t)bt * D_ + d] = acc;
    oacc[(size_t)bt * D_ + d] = acc;  // residual init
  }
  __syncthreads();
  if (t < E_) {
    float acc = router_b[t];
    for (int d = 0; d < D_; ++d) acc += xs[d] * router_w[d * E_ + t];
    float pd = (float)op - tile_centers[t];
    lg[t] = acc - pd * pd / SPREAD2X;
  }
  __syncthreads();
  if (t == 0) {
    int idx[K_]; float val[K_]; unsigned taken = 0;
    for (int k = 0; k < K_; ++k) {
      int bi = 0; float bv = -INFINITY;
      for (int e = 0; e < E_; ++e)
        if (!((taken >> e) & 1) && lg[e] > bv) { bv = lg[e]; bi = e; }
      idx[k] = bi; val[k] = bv; taken |= 1u << bi;
    }
    float mx = val[0], s = 0.f, g[K_];
    for (int k = 0; k < K_; ++k) { g[k] = expf(val[k] - mx); s += g[k]; }
    float go[E_];
    for (int e = 0; e < E_; ++e) go[e] = 0.f;
    for (int k = 0; k < K_; ++k) {
      float gv = g[k] / s;
      go[idx[k]] = gv;
      atomicAdd(&usage_out[idx[k]], gv);
      int p = atomicAdd(&cnt[idx[k]], 1);
      list[idx[k] * B + p] = bt;
      glist[idx[k] * B + p] = gv;
    }
    for (int e = 0; e < E_; ++e) gates_out[(size_t)bt * E_ + e] = go[e];
  }
}

// ================= kernel 3: gathered per-tile FFN via bf16 WMMA =================
#define LDS_BT 32768  // 512 rows x 64B: K-slab of W1t/W2t (swizzled), shared by all 8 waves
#define LDS_HB 65536  // 32 rows x 2048B: h buffer; xA (32 x 1024B) aliased here
// total 96KB static LDS: legal on gfx1250 (320KB/WGP); ~3 blocks/WGP occupancy

__global__ void __launch_bounds__(256)
trix_moe(const float* __restrict__ xbuf, const float* __restrict__ gauge,
         const float* __restrict__ theta,
         const __bf16* __restrict__ W1t, const __bf16* __restrict__ W2t,
         const float* __restrict__ b1, const float* __restrict__ b2,
         const int* __restrict__ cnt, const int* __restrict__ list,
         const float* __restrict__ glist, float* __restrict__ oacc, int B) {
  __shared__ __attribute__((aligned(16))) char smem[LDS_BT + LDS_HB];
  char* btb = smem;            // weight K-slab
  char* hb  = smem + LDS_BT;   // xA during GEMM1, then h (pre-rotation) for GEMM2

  int e = blockIdx.y;
  int base = blockIdx.x * MT;
  int count = cnt[e];
  if (base >= count) return;
  int nvalid = count - base; if (nvalid > MT) nvalid = MT;

  int t = threadIdx.x;
  int w = t >> 5;              // wave 0..7
  int wg = w >> 2;             // M-tile: rows wg*16 .. wg*16+15
  int wn = w & 3;              // col-group: 128 output cols
  int lane = t & 31;
  int laneHi = (lane >> 4) & 1;
  int lm = lane & 15;
  int mrow = wg * 16 + lm;     // A-fragment row for this lane

  // ---- stage A: xA[m][d] = bf16(x[tok] * gauge[e]) , swizzled rows of 1024B (32 rows)
  {
    const float4* ge4 = (const float4*)(gauge + e * D_);
#pragma unroll
    for (int j = 0; j < 16; ++j) {
      int lin = t + 256 * j;   // 4096 float4 units = 32 rows x 128
      int m = lin >> 7, c4 = lin & 127;
      if (m < nvalid) {
        int tokb = list[e * B + base + m];
        float4 xv = ((const float4*)(xbuf + (size_t)tokb * D_))[c4];
        float4 gv = ge4[c4];
        v4bf o = { (__bf16)(xv.x * gv.x), (__bf16)(xv.y * gv.y),
                   (__bf16)(xv.z * gv.z), (__bf16)(xv.w * gv.w) };
        *(v4bf*)swzp(hb, m, 1024, c4 * 8, 15) = o;
      }
    }
  }
  float ct = cosf(theta[e]), st = sinf(theta[e]);

  // ---- GEMM1: h = xA @ W1[e]  (32 x 1024, K=512), both H-halves held in registers
  v8f acc1[2][8];
#pragma unroll
  for (int hh = 0; hh < 2; ++hh)
#pragma unroll
    for (int nt = 0; nt < 8; ++nt) acc1[hh][nt] = (v8f)(0.0f);

  for (int hh = 0; hh < 2; ++hh) {
    for (int k0 = 0; k0 < D_; k0 += 32) {
      __syncthreads();  // previous slab fully consumed (also orders first use after xA fill)
      stage_bt(btb, W1t + ((size_t)e * H_ + hh * 512) * D_ + k0, D_);
      asm volatile("s_wait_asynccnt 0" ::: "memory");
      __syncthreads();
      ABFrag af;
      {
        int kb0 = (k0 + (laneHi ? 8 : 0)) * 2;
        af.h[0] = *(const v8bf*)swzp(hb, mrow, 1024, kb0, 15);
        af.h[1] = *(const v8bf*)swzp(hb, mrow, 1024, kb0 + 32, 15);
      }
#pragma unroll
      for (int nt = 0; nt < 8; ++nt) {
        int colLocal = wn * 128 + nt * 16 + lm;
        ABFrag bf;
        int kb = laneHi ? 16 : 0;
        bf.h[0] = *(const v8bf*)swzp(btb, colLocal, 64, kb, 3);
        bf.h[1] = *(const v8bf*)swzp(btb, colLocal, 64, kb + 32, 3);
        acc1[hh][nt] = __builtin_amdgcn_wmma_f32_16x16x32_bf16(
            false, af.v, false, bf.v, (short)0, acc1[hh][nt], false, false);
      }
    }
  }

  // ---- epilogue 1: bias + ReLU, write h (pre-rotation) to LDS (xA now dead)
  __syncthreads();
#pragma unroll
  for (int hh = 0; hh < 2; ++hh)
#pragma unroll
    for (int nt = 0; nt < 8; ++nt) {
      int n = hh * 512 + wn * 128 + nt * 16 + lm;
      float bias = b1[e * H_ + n];
#pragma unroll
      for (int r = 0; r < 8; ++r) {
        int m = wg * 16 + r + 8 * laneHi;
        float v = acc1[hh][nt][r] + bias;
        v = v > 0.f ? v : 0.f;
        *(__bf16*)swzp(hb, m, 2048, n * 2, 15) = (__bf16)v;
      }
    }

  // ---- GEMM2: y = rot(h) @ W2[e]  (32 x 512, K=1024); rotation fused into A-fragments
  v8f acc2[8];
#pragma unroll
  for (int nt = 0; nt < 8; ++nt) acc2[nt] = (v8f)(0.0f);

  for (int k0 = 0; k0 < H_; k0 += 32) {
    __syncthreads();
    stage_bt(btb, W2t + (size_t)e * D_ * H_ + k0, H_);
    asm volatile("s_wait_asynccnt 0" ::: "memory");
    __syncthreads();
    ABFrag af;
    {
      int kbase = k0 + (laneHi ? 8 : 0);
      float sgn = (k0 < 512) ? -st : st;      // rot_half: [-h2, h1]
      int partner = (k0 < 512) ? 512 : -512;
#pragma unroll
      for (int g = 0; g < 2; ++g) {
        int ke = kbase + g * 16;
        v8bf p = *(const v8bf*)swzp(hb, mrow, 2048, ke * 2, 15);
        v8bf q = *(const v8bf*)swzp(hb, mrow, 2048, (ke + partner) * 2, 15);
        v8bf o;
#pragma unroll
        for (int i = 0; i < 8; ++i)
          o[i] = (__bf16)((float)p[i] * ct + (float)q[i] * sgn);
        af.h[g] = o;
      }
    }
#pragma unroll
    for (int nt = 0; nt < 8; ++nt) {
      int colLocal = wn * 128 + nt * 16 + lm;
      ABFrag bf;
      int kb = laneHi ? 16 : 0;
      bf.h[0] = *(const v8bf*)swzp(btb, colLocal, 64, kb, 3);
      bf.h[1] = *(const v8bf*)swzp(btb, colLocal, 64, kb + 32, 3);
      acc2[nt] = __builtin_amdgcn_wmma_f32_16x16x32_bf16(
          false, af.v, false, bf.v, (short)0, acc2[nt], false, false);
    }
  }

  // ---- epilogue 2: out[tok] += gate * (y + b2)
#pragma unroll
  for (int nt = 0; nt < 8; ++nt) {
    int dcol = wn * 128 + nt * 16 + lm;
    float bias = b2[e * D_ + dcol];
#pragma unroll
    for (int r = 0; r < 8; ++r) {
      int m = wg * 16 + r + 8 * laneHi;
      if (m < nvalid) {
        int tokb = list[e * B + base + m];
        float gv = glist[e * B + base + m];
        atomicAdd(&oacc[(size_t)tokb * D_ + dcol], gv * (acc2[nt][r] + bias));
      }
    }
  }
}

// ================= kernel 4: head MLP 512->64->8 + sigmoid =================
__global__ void __launch_bounds__(64)
trix_head(const float* __restrict__ oacc, const float* __restrict__ h1_w,
          const float* __restrict__ h1_b, const float* __restrict__ h2_w,
          const float* __restrict__ h2_b, float* __restrict__ result, int B) {
  __shared__ float hid[64];
  int bt = blockIdx.x, t = threadIdx.x;
  const float* xr = oacc + (size_t)bt * D_;
  float acc = h1_b[t];
  for (int d = 0; d < D_; ++d) acc += xr[d] * h1_w[d * 64 + t];
  hid[t] = acc > 0.f ? acc : 0.f;
  __syncthreads();
  if (t < 8) {
    float z = h2_b[t];
    for (int j = 0; j < 64; ++j) z += hid[j] * h2_w[j * 8 + t];
    result[(size_t)bt * 8 + t] = 1.f / (1.f + expf(-z));
  }
}

extern "C" void kernel_launch(void* const* d_in, const int* in_sizes, int n_in,
                              void* d_out, int out_size, void* d_ws, size_t ws_size,
                              hipStream_t stream) {
  (void)n_in; (void)out_size; (void)ws_size;
  int B = in_sizes[0];
  const int*   op_idx       = (const int*)d_in[0];
  const int*   a            = (const int*)d_in[1];
  const int*   b            = (const int*)d_in[2];
  const int*   c            = (const int*)d_in[3];
  const float* op_embed_w   = (const float*)d_in[4];
  const float* proj_w       = (const float*)d_in[5];
  const float* proj_b       = (const float*)d_in[6];
  const float* router_w     = (const float*)d_in[7];
  const float* router_b     = (const float*)d_in[8];
  const float* tile_centers = (const float*)d_in[9];
  const float* gauge        = (const float*)d_in[10];
  const float* theta        = (const float*)d_in[11];
  const float* W1           = (const float*)d_in[12];
  const float* b1           = (const float*)d_in[13];
  const float* W2           = (const float*)d_in[14];
  const float* b2           = (const float*)d_in[15];
  const float* h1_w         = (const float*)d_in[16];
  const float* h1_b         = (const float*)d_in[17];
  const float* h2_w         = (const float*)d_in[18];
  const float* h2_b         = (const float*)d_in[19];

  float* out       = (float*)d_out;
  float* result    = out;                                  // [B,8]
  float* gates_out = out + (size_t)B * 8;                  // [B,16]
  float* usage_out = out + (size_t)B * 8 + (size_t)B * E_; // [16]

  char* ws = (char*)d_ws;
  size_t o = 0;
  __bf16* W1t  = (__bf16*)(ws + o); o += (size_t)E_ * H_ * D_ * 2;
  __bf16* W2t  = (__bf16*)(ws + o); o += (size_t)E_ * D_ * H_ * 2;
  float*  xbuf = (float*)(ws + o);  o += (size_t)B * D_ * 4;
  float*  oacc = (float*)(ws + o);  o += (size_t)B * D_ * 4;
  int*    cnt  = (int*)(ws + o);    o += 256;
  int*    list = (int*)(ws + o);    o += (size_t)E_ * B * 4;
  float*  glist= (float*)(ws + o);  o += (size_t)E_ * B * 4;

  trix_prep<<<16384, 256, 0, stream>>>(W1, W2, W1t, W2t, cnt, usage_out);
  trix_front<<<B, 128, 0, stream>>>(op_idx, a, b, c, op_embed_w, proj_w, proj_b,
                                    router_w, router_b, tile_centers,
                                    xbuf, oacc, gates_out, usage_out,
                                    cnt, list, glist, B);
  dim3 mg((B + MT - 1) / MT, E_);
  trix_moe<<<mg, 256, 0, stream>>>(xbuf, gauge, theta, W1t, W2t, b1, b2,
                                   cnt, list, glist, oacc, B);
  trix_head<<<B, 64, 0, stream>>>(oacc, h1_w, h1_b, h2_w, h2_b, result, B);
}
